// PAMAttentionBlock__21088289424048
// MI455X (gfx1250) — matrix operand order repaired and checked by reading, not compiled, not run on hardware
//
#include <hip/hip_runtime.h>
#include <hip/hip_bf16.h>

typedef unsigned short u16;
typedef unsigned int   u32;

typedef __attribute__((ext_vector_type(16))) __bf16 v16bf;
typedef __attribute__((ext_vector_type(8)))  float  v8f;

#define B_    4
#define C_IN  128
#define H_    96
#define W_    320
#define HW_   (H_*W_)      /* 30720 */
#define EPS_  1e-5f

/* ---------------- workspace layout (bytes) ---------------- */
#define OFF_BIAS  180224u
#define OFF_FA    184320u
#define FEATB_    15728640u
#define OFF_FB    (OFF_FA + FEATB_)
#define OFF_FX    (OFF_FB + FEATB_)
#define OFF_FY    (OFF_FX + FEATB_)
#define OFF_Q     (OFF_FY + FEATB_)
#define OFF_KB    (OFF_Q  + 7864320u)
#define OFF_VB    (OFF_KB + 7864320u)
/* ushort offsets of folded bf16 weights inside ws */
#define WOFF_W1   0
#define WOFF_W2   8192
#define WOFF_W3   45056
#define WOFF_WQ   81920
#define WOFF_WK   83968
#define WOFF_WV   86016

/* ---------------- helpers ---------------- */
__device__ __forceinline__ u16 f2bf(float f) {
  u32 u = __float_as_uint(f);
  u32 r = u + 0x7FFFu + ((u >> 16) & 1u);
  return (u16)(r >> 16);
}
__device__ __forceinline__ u32 pack2bf(float a, float b) {
  return (u32)f2bf(a) | ((u32)f2bf(b) << 16);
}

union ABFrag { v16bf v; u32 u[8]; };

/* A matrix 16x32 bf16, stored [M][K] row-major, ld elements per row.
   ISA layout: lanes 0-15 hold K pairs {0..7,16..23}, lanes 16-31 {8..15,24..31}. */
__device__ __forceinline__ v16bf load_a_frag(const u16* base, int ld, int lane) {
  ABFrag f;
  int m  = lane & 15;
  int kb = (lane & 16) ? 8 : 0;
  const u16* row = base + m * ld;
#pragma unroll
  for (int i = 0; i < 4; ++i) {
    f.u[i]     = *(const u32*)(row + kb + 2*i);
    f.u[i + 4] = *(const u32*)(row + 16 + kb + 2*i);
  }
  return f.v;
}

/* B matrix 32x16 (KxN) bf16, stored transposed [N][K], ld elements per N-row.
   Lanes 0-15: K={0..7,16..23}; lanes 16-31: K={8..15,24..31}. */
__device__ __forceinline__ v16bf load_b_frag(const u16* base, int ld, int lane) {
  ABFrag f;
  int n  = lane & 15;
  int kb = (lane & 16) ? 8 : 0;
  const u16* row = base + n * ld;
#pragma unroll
  for (int i = 0; i < 4; ++i) {
    f.u[i]     = *(const u32*)(row + kb + 2*i);
    f.u[i + 4] = *(const u32*)(row + 16 + kb + 2*i);
  }
  return f.v;
}

__device__ __forceinline__ v8f wmma_bf(v16bf a, v16bf b, v8f c) {
  return __builtin_amdgcn_wmma_f32_16x16x32_bf16(false, a, false, b, (short)0, c, false, false);
}

/* ---------------- BN-fold / weight prep ---------------- */
struct PP { const float* p[36]; };   /* 6 layers x (w,b,g,beta,m,v) */

__global__ void prep_kernel(PP pp, u16* __restrict__ wsu, float* __restrict__ bout) {
  int tid = blockIdx.x * blockDim.x + threadIdx.x;
  int nth = gridDim.x * blockDim.x;
  const int couts[6] = {64, 64, 64, 32, 32, 64};
  const int boff[6]  = {0, 64, 128, 192, 224, 256};
  for (int l = 0; l < 6; ++l) {
    const float* bb = pp.p[6*l + 1];
    const float* g  = pp.p[6*l + 2];
    const float* be = pp.p[6*l + 3];
    const float* mm = pp.p[6*l + 4];
    const float* vv = pp.p[6*l + 5];
    for (int c = tid; c < couts[l]; c += nth) {
      float sc = g[c] * rsqrtf(vv[c] + EPS_);
      bout[boff[l] + c] = (bb[c] - mm[c]) * sc + be[c];
    }
  }
  {
    const float* w = pp.p[0]; const float* g = pp.p[2]; const float* vv = pp.p[5];
    for (int i = tid; i < 64*128; i += nth) {
      int co = i >> 7;
      wsu[WOFF_W1 + i] = f2bf(w[i] * g[co] * rsqrtf(vv[co] + EPS_));
    }
  }
  for (int l = 1; l <= 2; ++l) {
    const float* w = pp.p[6*l]; const float* g = pp.p[6*l+2]; const float* vv = pp.p[6*l+5];
    u16* dst = wsu + WOFF_W2 + (l - 1) * 36864;
    for (int i = tid; i < 64*576; i += nth) {
      int co = i / 576, rem = i % 576, t9 = rem >> 6, ci = rem & 63;
      float sc = g[co] * rsqrtf(vv[co] + EPS_);
      dst[i] = f2bf(w[(co*64 + ci)*9 + t9] * sc);
    }
  }
  {
    const int lyr[3]  = {3, 4, 5};
    const int cnt[3]  = {2048, 2048, 4096};
    const int doff[3] = {WOFF_WQ, WOFF_WK, WOFF_WV};
    for (int j = 0; j < 3; ++j) {
      int l = lyr[j];
      const float* w = pp.p[6*l]; const float* g = pp.p[6*l+2]; const float* vv = pp.p[6*l+5];
      for (int i = tid; i < cnt[j]; i += nth) {
        int co = i >> 6;
        wsu[doff[j] + i] = f2bf(w[i] * g[co] * rsqrtf(vv[co] + EPS_));
      }
    }
  }
}

/* ---------------- head1: 1x1 conv 128->64, NCHW f32 in, channels-last bf16 out;
   weights in LDS, wave register-blocks all 4 co-tiles over one shared B fragment ---------------- */
__global__ __launch_bounds__(256) void head1_kernel(const float* __restrict__ x,
    const u16* __restrict__ w1, const float* __restrict__ bias, u16* __restrict__ out) {
  __shared__ __attribute__((aligned(16))) u16 xT[128 * 130];
  __shared__ __attribute__((aligned(16))) u16 wL[64 * 128];
  int bid = blockIdx.x;
  int b = bid / 240, tile = bid % 240;
  int pixbase = tile * 128;
  const float* xb = x + (size_t)b * C_IN * HW_ + pixbase;
  int tid = threadIdx.x;
  for (int i = tid; i < 4096; i += 256)           /* stage 16 KB weights */
    ((u32*)wL)[i] = ((const u32*)w1)[i];
  for (int i = tid; i < 128 * 128; i += 256) {
    int ch = i >> 7, pix = i & 127;
    xT[pix * 130 + ch] = f2bf(xb[(size_t)ch * HW_ + pix]);
  }
  __syncthreads();
  int wave = tid >> 5, lane = tid & 31;
  v8f acc[4] = {};
#pragma unroll
  for (int kk = 0; kk < 128; kk += 32) {
    v16bf bb = load_b_frag(&xT[wave * 16 * 130 + kk], 130, lane);
#pragma unroll
    for (int c = 0; c < 4; ++c) {
      v16bf a = load_a_frag(wL + c * 16 * 128 + kk, 128, lane);
      acc[c] = wmma_bf(a, bb, acc[c]);
    }
  }
  int n = lane & 15, mb = (lane & 16) ? 8 : 0;
  int pix = wave * 16 + n;
#pragma unroll
  for (int c = 0; c < 4; ++c) {
    u32 pk[4];
#pragma unroll
    for (int r = 0; r < 8; r += 2) {
      int co = c * 16 + mb + r;
      float v0 = acc[c][r] + bias[co];
      float v1 = acc[c][r + 1] + bias[co + 1];
      v0 = fmaxf(v0, 0.1f * v0); v1 = fmaxf(v1, 0.1f * v1);
      pk[r >> 1] = pack2bf(v0, v1);
    }
    u32* op = (u32*)(out + ((size_t)b * HW_ + pixbase + pix) * 64 + c * 16 + mb);
    *(uint4*)op = make_uint4(pk[0], pk[1], pk[2], pk[3]);
  }
}

/* ---------------- 3x3 conv 64->64 (+LeakyReLU): implicit GEMM, weights+halo in dynamic LDS,
   block covers 2 output rows x 64 cols; each wave register-blocks 2co x 2row tiles ---------------- */
#define SMEM_W3   73728                      /* 64*576 bf16 weights  */
#define SMEM_IN3  (4*66*64*2)                /* 4-row halo           */
#define SMEM_3X3  (SMEM_W3 + SMEM_IN3)       /* 107520 bytes         */

__global__ __launch_bounds__(256) void conv3x3_kernel(const u16* __restrict__ fin,
    const u16* __restrict__ wf, const float* __restrict__ bias, u16* __restrict__ fout) {
  extern __shared__ __attribute__((aligned(16))) char smem3[];
  u16* wL  = (u16*)smem3;
  u16* inL = (u16*)(smem3 + SMEM_W3);
  int bid = blockIdx.x;
  int b = bid / 240;
  int rem = bid % 240;
  int hblk = rem / 5, wt = rem % 5;
  int h0 = hblk * 2;
  int w0 = wt * 64;
  int tid = threadIdx.x;
  for (int i = tid; i < 18432; i += 256)
    ((u32*)wL)[i] = ((const u32*)wf)[i];
  for (int i = tid; i < 4 * 66 * 32; i += 256) {
    int cu  = i & 31;
    int col = (i >> 5) % 66;
    int r   = i / (66 * 32);
    int hh = h0 + r - 1, ww = w0 + col - 1;
    u32 val = 0;
    if (hh >= 0 && hh < H_ && ww >= 0 && ww < W_)
      val = *(const u32*)(fin + (((size_t)b * H_ + hh) * W_ + ww) * 64 + cu * 2);
    *(u32*)(&inL[(r * 66 + col) * 64 + cu * 2]) = val;
  }
  __syncthreads();
  int wave = tid >> 5, lane = tid & 31;
  int cw = wave >> 2;        /* co16 pair: {2cw, 2cw+1} */
  int pw = wave & 3;         /* pixel 16-tile within 64 */
  v8f acc00 = {}, acc01 = {}, acc10 = {}, acc11 = {};
#pragma unroll
  for (int t9 = 0; t9 < 9; ++t9) {
    int dy = t9 / 3, dx = t9 % 3;
#pragma unroll
    for (int kk = 0; kk < 64; kk += 32) {
      v16bf a0 = load_a_frag(wL + ((cw * 2 + 0) * 16) * 576 + t9 * 64 + kk, 576, lane);
      v16bf a1 = load_a_frag(wL + ((cw * 2 + 1) * 16) * 576 + t9 * 64 + kk, 576, lane);
      v16bf b0 = load_b_frag(&inL[((0 + dy) * 66 + pw * 16 + dx) * 64 + kk], 64, lane);
      v16bf b1 = load_b_frag(&inL[((1 + dy) * 66 + pw * 16 + dx) * 64 + kk], 64, lane);
      acc00 = wmma_bf(a0, b0, acc00);
      acc01 = wmma_bf(a0, b1, acc01);
      acc10 = wmma_bf(a1, b0, acc10);
      acc11 = wmma_bf(a1, b1, acc11);
    }
  }
  int n = lane & 15, mb = (lane & 16) ? 8 : 0;
  int pix = pw * 16 + n;
#pragma unroll
  for (int c = 0; c < 2; ++c) {
#pragma unroll
    for (int rr = 0; rr < 2; ++rr) {
      v8f acc = (c == 0) ? (rr == 0 ? acc00 : acc01) : (rr == 0 ? acc10 : acc11);
      int co16 = cw * 2 + c;
      int h = h0 + rr;
      u32 pk[4];
#pragma unroll
      for (int r = 0; r < 8; r += 2) {
        int co = co16 * 16 + mb + r;
        float v0 = acc[r] + bias[co];
        float v1 = acc[r + 1] + bias[co + 1];
        v0 = fmaxf(v0, 0.1f * v0); v1 = fmaxf(v1, 0.1f * v1);
        pk[r >> 1] = pack2bf(v0, v1);
      }
      u32* op = (u32*)(fout + (((size_t)b * H_ + h) * W_ + w0 + pix) * 64 + co16 * 16 + mb);
      *(uint4*)op = make_uint4(pk[0], pk[1], pk[2], pk[3]);
    }
  }
}

/* ---------------- 1x1 conv 64->COUT (no act): Q/K/V; weights in LDS,
   wave register-blocks all CT co-tiles over one shared B fragment ---------------- */
template <int COUT>
__global__ __launch_bounds__(256) void conv1x1_kernel(const u16* __restrict__ fin,
    const u16* __restrict__ wq, const float* __restrict__ bias, u16* __restrict__ out) {
  __shared__ __attribute__((aligned(16))) u16 wL[COUT * 64];
  int bid = blockIdx.x;
  int b = bid / 240, tile = bid % 240;
  size_t pixbase = (size_t)b * HW_ + tile * 128;
  int tid = threadIdx.x, wave = tid >> 5, lane = tid & 31;
  const int CT = COUT / 16;
  for (int i = tid; i < COUT * 32; i += 256)
    ((u32*)wL)[i] = ((const u32*)wq)[i];
  /* prefetch the next pixel tile while we compute this one */
  if (tile < 239)
    __builtin_prefetch(fin + (pixbase + 128) * 64 + tid * 32, 0, 1);
  __syncthreads();
  v8f acc[CT] = {};
  const u16* bbase = fin + (pixbase + wave * 16) * 64;
#pragma unroll
  for (int kk = 0; kk < 64; kk += 32) {
    v16bf bb = load_b_frag(bbase + kk, 64, lane);
#pragma unroll
    for (int j = 0; j < CT; ++j) {
      v16bf a = load_a_frag(wL + j * 16 * 64 + kk, 64, lane);
      acc[j] = wmma_bf(a, bb, acc[j]);
    }
  }
  int n = lane & 15, mb = (lane & 16) ? 8 : 0;
#pragma unroll
  for (int j = 0; j < CT; ++j) {
    u32 pk[4];
#pragma unroll
    for (int r = 0; r < 8; r += 2) {
      int co = j * 16 + mb + r;
      pk[r >> 1] = pack2bf(acc[j][r] + bias[co], acc[j][r + 1] + bias[co + 1]);
    }
    u32* op = (u32*)(out + (pixbase + wave * 16 + n) * COUT + j * 16 + mb);
    *(uint4*)op = make_uint4(pk[0], pk[1], pk[2], pk[3]);
  }
}

/* ---------------- row attention: one block per (b,h), seq = W = 320, dynamic LDS ----------------
   layout: Kl[320*32]u16 @0 | Vt[64*320]u16 @20480 | Ql[32*32]u16 @61440 |
           Sf[32*320]f32 @63488 | Pl[32*320]u16 @104448 | red[256]f32 @124928  => 125952 B */
#define SMEM_RA 125952

__global__ __launch_bounds__(256) void row_attn_kernel(const u16* __restrict__ Qg,
    const u16* __restrict__ Kg, const u16* __restrict__ Vg, float* __restrict__ out) {
  extern __shared__ __attribute__((aligned(16))) char smemA[];
  u16*   Kl  = (u16*)smemA;
  u16*   Vt  = (u16*)(smemA + 20480);
  u16*   Ql  = (u16*)(smemA + 61440);
  float* Sf  = (float*)(smemA + 63488);
  u16*   Pl  = (u16*)(smemA + 104448);
  float* red = (float*)(smemA + 124928);
  int bid = blockIdx.x;
  int b = bid / H_, h = bid % H_;
  size_t rowpix = (size_t)(b * H_ + h) * W_;
  int tid = threadIdx.x, wave = tid >> 5, lane = tid & 31;
  const float scale = 0.17677669529663687f;   /* 32^-0.5 */
  for (int i = tid; i < 320 * 16; i += 256)
    ((u32*)Kl)[i] = ((const u32*)(Kg + rowpix * 32))[i];
  for (int i = tid; i < 320 * 64; i += 256) {
    int vc = i & 63, key = i >> 6;
    Vt[vc * 320 + key] = Vg[rowpix * 64 + i];
  }
  for (int qi = 0; qi < 10; ++qi) {
    __syncthreads();
    for (int i = tid; i < 512; i += 256)
      ((u32*)Ql)[i] = ((const u32*)(Qg + (rowpix + qi * 32) * 32))[i];
    /* prefetch next Q tile (gfx1250 global_prefetch) */
    if (qi < 9)
      __builtin_prefetch(Qg + (rowpix + (qi + 1) * 32) * 32 + tid * 4, 0, 1);
    __syncthreads();
    v16bf aQ0 = load_a_frag(Ql, 32, lane);
    v16bf aQ1 = load_a_frag(Ql + 16 * 32, 32, lane);
    for (int t = wave; t < 40; t += 8) {          /* S = Q*K^T : one WMMA per tile */
      int qt = t / 20, kt = t % 20;
      v16bf bb = load_b_frag(Kl + kt * 16 * 32, 32, lane);
      v8f acc = {};
      acc = wmma_bf(qt ? aQ1 : aQ0, bb, acc);
      int n = lane & 15, mb = (lane & 16) ? 8 : 0;
#pragma unroll
      for (int r = 0; r < 8; ++r)
        Sf[(qt * 16 + mb + r) * 320 + kt * 16 + n] = acc[r] * scale;
    }
    __syncthreads();
    /* softmax over 320: 32 rows x 8 threads */
    {
      int row = tid >> 3, seg = tid & 7;
      const float* srow = &Sf[row * 320 + seg * 40];
      float mx = -1e30f;
      for (int jj = 0; jj < 40; ++jj) mx = fmaxf(mx, srow[jj]);
      red[tid] = mx;
      __syncthreads();
      float rmx = -1e30f;
      for (int jj = 0; jj < 8; ++jj) rmx = fmaxf(rmx, red[(row << 3) + jj]);
      __syncthreads();
      float sum = 0.f;
      for (int jj = 0; jj < 40; ++jj) sum += __expf(srow[jj] - rmx);
      red[tid] = sum;
      __syncthreads();
      float rs = 0.f;
      for (int jj = 0; jj < 8; ++jj) rs += red[(row << 3) + jj];
      float inv = 1.0f / rs;
      u16* prow = &Pl[row * 320 + seg * 40];
      for (int jj = 0; jj < 40; ++jj) prow[jj] = f2bf(__expf(srow[jj] - rmx) * inv);
    }
    __syncthreads();
    /* ctx = P * V : 8 tiles (2 q x 4 vc), one per wave */
    {
      int qt = wave >> 2, vt = wave & 3;
      v8f acc = {};
#pragma unroll
      for (int kk = 0; kk < 320; kk += 32) {
        v16bf a  = load_a_frag(Pl + qt * 16 * 320 + kk, 320, lane);
        v16bf bb = load_b_frag(Vt + (vt * 16) * 320 + kk, 320, lane);
        acc = wmma_bf(a, bb, acc);
      }
      int n = lane & 15, mb = (lane & 16) ? 8 : 0;
      int vc = vt * 16 + n;
      float* op = out + (((size_t)b * 64 + vc) * H_ + h) * W_ + qi * 32 + qt * 16 + mb;
      *(float4*)op       = make_float4(acc[0], acc[1], acc[2], acc[3]);
      *(float4*)(op + 4) = make_float4(acc[4], acc[5], acc[6], acc[7]);
    }
  }
}

/* ---------------- column attention: one block per (b,w), seq = H = 96 ---------------- */
__global__ __launch_bounds__(256) void col_attn_kernel(const u16* __restrict__ Qg,
    const u16* __restrict__ Kg, const u16* __restrict__ Vg, float* __restrict__ out) {
  __shared__ __attribute__((aligned(16))) u16   Kl[96 * 32];
  __shared__ __attribute__((aligned(16))) u16   Vt[64 * 96];
  __shared__ __attribute__((aligned(16))) u16   Ql[48 * 32];
  __shared__ __attribute__((aligned(16))) float Sf[48 * 96];
  __shared__ __attribute__((aligned(16))) u16   Pl[48 * 96];
  __shared__ float red[192];
  int bid = blockIdx.x;
  int b = bid / W_, w = bid % W_;
  int tid = threadIdx.x, wave = tid >> 5, lane = tid & 31;
  const float scale = 0.17677669529663687f;
  for (int i = tid; i < 96 * 16; i += 256) {
    int hh = i >> 4, cu = i & 15;
    ((u32*)Kl)[i] = *(const u32*)(Kg + (((size_t)b * H_ + hh) * W_ + w) * 32 + cu * 2);
  }
  for (int i = tid; i < 96 * 64; i += 256) {
    int vc = i & 63, hh = i >> 6;
    Vt[vc * 96 + hh] = Vg[(((size_t)b * H_ + hh) * W_ + w) * 64 + vc];
  }
  for (int qi = 0; qi < 2; ++qi) {
    __syncthreads();
    for (int i = tid; i < 48 * 16; i += 256) {
      int hh = qi * 48 + (i >> 4), cu = i & 15;
      ((u32*)Ql)[i] = *(const u32*)(Qg + (((size_t)b * H_ + hh) * W_ + w) * 32 + cu * 2);
    }
    __syncthreads();
    for (int t = wave; t < 18; t += 8) {          /* S tiles: 3 x 6 */
      int qt = t / 6, kt = t % 6;
      v16bf a  = load_a_frag(Ql + qt * 16 * 32, 32, lane);
      v16bf bb = load_b_frag(Kl + kt * 16 * 32, 32, lane);
      v8f acc = {};
      acc = wmma_bf(a, bb, acc);
      int n = lane & 15, mb = (lane & 16) ? 8 : 0;
#pragma unroll
      for (int r = 0; r < 8; ++r)
        Sf[(qt * 16 + mb + r) * 96 + kt * 16 + n] = acc[r] * scale;
    }
    __syncthreads();
    /* softmax over 96: 48 rows x 4 threads */
    int row = tid >> 2, seg = tid & 3;
    const float* srow = &Sf[row * 96 + seg * 24];
    if (tid < 192) {
      float mx = -1e30f;
      for (int jj = 0; jj < 24; ++jj) mx = fmaxf(mx, srow[jj]);
      red[tid] = mx;
    }
    __syncthreads();
    float rmx = 0.f;
    if (tid < 192)
      rmx = fmaxf(fmaxf(red[row * 4], red[row * 4 + 1]), fmaxf(red[row * 4 + 2], red[row * 4 + 3]));
    __syncthreads();
    if (tid < 192) {
      float sum = 0.f;
      for (int jj = 0; jj < 24; ++jj) sum += __expf(srow[jj] - rmx);
      red[tid] = sum;
    }
    __syncthreads();
    if (tid < 192) {
      float rs = red[row * 4] + red[row * 4 + 1] + red[row * 4 + 2] + red[row * 4 + 3];
      float inv = 1.0f / rs;
      u16* prow = &Pl[row * 96 + seg * 24];
      for (int jj = 0; jj < 24; ++jj) prow[jj] = f2bf(__expf(srow[jj] - rmx) * inv);
    }
    __syncthreads();
    for (int t = wave; t < 12; t += 8) {          /* ctx tiles: 3 x 4 */
      int qt = t >> 2, vt = t & 3;
      v8f acc = {};
#pragma unroll
      for (int kk = 0; kk < 96; kk += 32) {
        v16bf a  = load_a_frag(Pl + qt * 16 * 96 + kk, 96, lane);
        v16bf bb = load_b_frag(Vt + (vt * 16) * 96 + kk, 96, lane);
        acc = wmma_bf(a, bb, acc);
      }
      int n = lane & 15, mb = (lane & 16) ? 8 : 0;
      int vc = vt * 16 + n;
      int q0 = qi * 48 + qt * 16 + mb;
      float* op = out + (((size_t)b * 64 + vc) * H_ + q0) * W_ + w;
#pragma unroll
      for (int r = 0; r < 8; ++r) op[(size_t)r * W_] += acc[r];
    }
  }
}

/* ---------------- launch ---------------- */
extern "C" void kernel_launch(void* const* d_in, const int* in_sizes, int n_in,
                              void* d_out, int out_size, void* d_ws, size_t ws_size,
                              hipStream_t stream) {
  (void)in_sizes; (void)n_in; (void)out_size; (void)ws_size;
  const float* x = (const float*)d_in[0];
  const float* y = (const float*)d_in[1];
  PP pp;
  for (int i = 0; i < 36; ++i) pp.p[i] = (const float*)d_in[2 + i];

  char* ws = (char*)d_ws;
  u16*   wsu  = (u16*)ws;
  float* bias = (float*)(ws + OFF_BIAS);
  u16* fa = (u16*)(ws + OFF_FA);
  u16* fb = (u16*)(ws + OFF_FB);
  u16* fx = (u16*)(ws + OFF_FX);
  u16* fy = (u16*)(ws + OFF_FY);
  u16* Qb = (u16*)(ws + OFF_Q);
  u16* Kb = (u16*)(ws + OFF_KB);
  u16* Vb = (u16*)(ws + OFF_VB);
  float* out = (float*)d_out;

  prep_kernel<<<64, 256, 0, stream>>>(pp, wsu, bias);

  head1_kernel<<<960, 256, 0, stream>>>(x, wsu + WOFF_W1, bias + 0, fa);
  conv3x3_kernel<<<960, 256, SMEM_3X3, stream>>>(fa, wsu + WOFF_W2, bias + 64, fb);
  conv3x3_kernel<<<960, 256, SMEM_3X3, stream>>>(fb, wsu + WOFF_W3, bias + 128, fx);

  head1_kernel<<<960, 256, 0, stream>>>(y, wsu + WOFF_W1, bias + 0, fa);
  conv3x3_kernel<<<960, 256, SMEM_3X3, stream>>>(fa, wsu + WOFF_W2, bias + 64, fb);
  conv3x3_kernel<<<960, 256, SMEM_3X3, stream>>>(fb, wsu + WOFF_W3, bias + 128, fy);

  conv1x1_kernel<32><<<960, 256, 0, stream>>>(fx, wsu + WOFF_WQ, bias + 192, Qb);
  conv1x1_kernel<32><<<960, 256, 0, stream>>>(fy, wsu + WOFF_WK, bias + 224, Kb);
  conv1x1_kernel<64><<<960, 256, 0, stream>>>(fy, wsu + WOFF_WV, bias + 256, Vb);

  row_attn_kernel<<<B_ * H_, 256, SMEM_RA, stream>>>(Qb, Kb, Vb, out);
  col_attn_kernel<<<B_ * W_, 256, 0, stream>>>(Qb, Kb, Vb, out);
}
